// AggrHGraphConvWindow_19808389169874
// MI455X (gfx1250) — matrix operand order repaired and checked by reading, not compiled, not run on hardware
//
#include <hip/hip_runtime.h>
#include <math.h>

typedef __attribute__((ext_vector_type(16))) _Float16 v16h;
typedef __attribute__((ext_vector_type(8)))  float    v8f;

#define N_NODE 500
#define N_POD  20000
#define N_SVC  2000
#define TT     16
#define FF     64
#define HH     128
#define OO     64
#define N_TOT  22500

// ---------------------------------------------------------------------------
// Degree counting + rsqrt. unsafeAtomicAdd -> native global_atomic_add_f32.
// ---------------------------------------------------------------------------
__global__ void degree_count_k(const int* __restrict__ src, const int* __restrict__ dst,
                               float* __restrict__ outd, float* __restrict__ ind, int E) {
    int e = blockIdx.x * blockDim.x + threadIdx.x;
    if (e < E) {
        unsafeAtomicAdd(outd + src[e], 1.0f);
        unsafeAtomicAdd(ind  + dst[e], 1.0f);
    }
}

__global__ void to_rinv_k(float* a, int n) {
    int i = blockIdx.x * blockDim.x + threadIdx.x;
    if (i < n) {
        float d = a[i];
        if (d < 1.0f) d = 1.0f;
        a[i] = rsqrtf(d);
    }
}

__global__ void add_bias_k(const float* a, const float* b, float* o, int n) {
    int i = blockIdx.x * blockDim.x + threadIdx.x;
    if (i < n) o[i] = a[i] + b[i];
}

// ---------------------------------------------------------------------------
// Pack weight matrix into WMMA B-operand fragment order (f16).
// Fragment layout per (kt, nt): 32 lanes x 16 halves, element j of lane L:
//   n = nt*16 + (L & 15),  k = kt*32 + (j & 7) + 8*(L >> 4) + 16*(j >> 3)   (ISA B layout)
// transpose=0: B[k][n] = W[k*N + n]; transpose=1: B[k][n] = W[n*K + k]
// ---------------------------------------------------------------------------
__global__ void pack_b16_k(const float* __restrict__ W, _Float16* __restrict__ out,
                           int K, int N, int transpose) {
    int tid = blockIdx.x * blockDim.x + threadIdx.x;
    int total = (K / 32) * (N / 16) * 512;
    if (tid >= total) return;
    int j    = tid & 15;
    int lane = (tid >> 4) & 31;
    int frag = tid >> 9;
    int ntiles = N / 16;
    int nt = frag % ntiles;
    int kt = frag / ntiles;
    int hi = lane >> 4;
    int n  = nt * 16 + (lane & 15);
    int k  = kt * 32 + (j & 7) + 8 * hi + 16 * (j >> 3);
    float v = transpose ? W[(size_t)n * K + k] : W[(size_t)k * N + n];
    out[tid] = (_Float16)v;
}

// ---------------------------------------------------------------------------
// Edge scatter: m[dst] += x[src] * rinv_outdeg[src]   (T*F = 1024 floats/edge)
// One block (256 threads) per edge, 4 floats per thread, native f32 atomics.
// ---------------------------------------------------------------------------
__global__ void scatter_add_k(const float* __restrict__ x, const int* __restrict__ src,
                              const int* __restrict__ dst, const float* __restrict__ rv,
                              float* __restrict__ m) {
    int e = blockIdx.x;
    int s = src[e], d = dst[e];
    float sc = rv[s];
    const float4* xr = (const float4*)(x + (size_t)s * 1024);
    float4 v = xr[threadIdx.x];
    float* mr = m + (size_t)d * 1024 + threadIdx.x * 4;
    unsafeAtomicAdd(mr + 0, v.x * sc);
    unsafeAtomicAdd(mr + 1, v.y * sc);
    unsafeAtomicAdd(mr + 2, v.z * sc);
    unsafeAtomicAdd(mr + 3, v.w * sc);
}

// ---------------------------------------------------------------------------
// Per-dst-type combine: for each relation r: y += (m_r * rsqrt(ind_r)) @ W_r
// then (y + sum biases) * inv_div, ReLU, write feat row-block.
// One wave owns a 16-row M tile and ALL 8 N-tiles (H=128): A fragment is
// built once per (rel, kt) and reused for 8 back-to-back WMMAs.
// A fragment (16-bit A 16x32 layout): element j of lane L (m = L&15, hi=L>>4):
//   k = (j%8) + 8*(2*(j/8) + hi)
// ---------------------------------------------------------------------------
__global__ void feat_combine_k(
    const float* m0, const float* rv0, const _Float16* w0, const float* b0,
    const float* m1, const float* rv1, const _Float16* w1, const float* b1,
    const float* m2, const float* rv2, const _Float16* w2, const float* b2,
    int nrel, float inv_div, float* __restrict__ outbase) {

    const int lane  = threadIdx.x;
    const int mtile = blockIdx.x;
    const int hi = lane >> 4;
    const int lm = lane & 15;
    const int row  = mtile * 16 + lm;    // (node*T + t)
    const int node = row >> 4;

    const float*    ms[3]  = { m0, m1, m2 };
    const float*    rvs[3] = { rv0, rv1, rv2 };
    const _Float16* pws[3] = { w0, w1, w2 };

    v8f acc[8];
#pragma unroll
    for (int nt = 0; nt < 8; ++nt) acc[nt] = (v8f){};

    for (int rel = 0; rel < nrel; ++rel) {
        const float* ar = ms[rel] + (size_t)row * FF;
        float sc = rvs[rel][node];
#pragma unroll
        for (int kt = 0; kt < 2; ++kt) {
            const float* p = ar + kt * 32 + 8 * hi;
            v16h a;
#pragma unroll
            for (int jj = 0; jj < 8; ++jj) {
                a[jj]     = (_Float16)(p[jj]      * sc);   // k = jj + 8*hi
                a[8 + jj] = (_Float16)(p[16 + jj] * sc);   // k = jj + 16 + 8*hi
            }
            const _Float16* wb = pws[rel] + (size_t)kt * 8 * 512 + lane * 16;
#pragma unroll
            for (int nt = 0; nt < 8; ++nt) {
                const v16h* bw = (const v16h*)(wb + (size_t)nt * 512);
                acc[nt] = __builtin_amdgcn_wmma_f32_16x16x32_f16(false, a, false, *bw,
                                                                 (short)0, acc[nt], false, false);
            }
        }
    }
#pragma unroll
    for (int nt = 0; nt < 8; ++nt) {
        const int n = nt * 16 + lm;
        float bs = b0[n];
        if (nrel > 1) bs += b1[n];
        if (nrel > 2) bs += b2[n];
#pragma unroll
        for (int r = 0; r < 8; ++r) {
            int rowg = mtile * 16 + r + 8 * hi;   // D layout: lanes16-31 hold M = r+8
            float v = (acc[nt][r] + bs) * inv_div;
            outbase[(size_t)rowg * HH + n] = v > 0.0f ? v : 0.0f;
        }
    }
}

// ---------------------------------------------------------------------------
// One LSTM layer, WMMA recurrence. One wave owns 16 batch rows for all 16
// timesteps; h/c live in LDS; gate GEMM [16 x KIN]x[KIN x 256] + [16x64]x[64x256]
// done with v_wmma_f32_16x16x32_f16 per 16-wide N tile.
// ---------------------------------------------------------------------------
template <int KIN>
__global__ void lstm_layer_k(const float* __restrict__ x, const _Float16* __restrict__ wih,
                             const _Float16* __restrict__ whh, const float* __restrict__ bias,
                             float* __restrict__ out, int Bn, int relu_out) {
    __shared__ float gates[16][4 * OO];   // 16 x 256
    __shared__ float hbuf[16][OO];
    __shared__ float cbuf[16][OO];

    const int lane = threadIdx.x;
    const int hi = lane >> 4;
    const int lm = lane & 15;
    const int base = blockIdx.x * 16;
    const int node = base + lm;
    const bool valid = node < Bn;

    for (int i = lane; i < 16 * OO; i += 32) {
        (&hbuf[0][0])[i] = 0.0f;
        (&cbuf[0][0])[i] = 0.0f;
    }
    __syncthreads();

    for (int t = 0; t < TT; ++t) {
        // A fragments from the layer input (f32 -> f16)
        v16h ax[KIN / 32];
        const float* xr = x + ((size_t)node * TT + t) * KIN;
#pragma unroll
        for (int kt = 0; kt < KIN / 32; ++kt) {
#pragma unroll
            for (int jj = 0; jj < 8; ++jj) {
                float v0 = valid ? xr[kt * 32 + 8 * hi + jj]      : 0.0f;
                float v1 = valid ? xr[kt * 32 + 16 + 8 * hi + jj] : 0.0f;
                ax[kt][jj]     = (_Float16)v0;
                ax[kt][8 + jj] = (_Float16)v1;
            }
        }
        // A fragments from recurrent state h (LDS)
        v16h ah[2];
#pragma unroll
        for (int kt = 0; kt < 2; ++kt) {
#pragma unroll
            for (int jj = 0; jj < 8; ++jj) {
                ah[kt][jj]     = (_Float16)hbuf[lm][kt * 32 + 8 * hi + jj];
                ah[kt][8 + jj] = (_Float16)hbuf[lm][kt * 32 + 16 + 8 * hi + jj];
            }
        }
        // gate GEMM: 16 N-tiles of 16 columns (N = 4*O = 256)
        for (int nt = 0; nt < 16; ++nt) {
            v8f acc = {};
#pragma unroll
            for (int kt = 0; kt < KIN / 32; ++kt) {
                const v16h* bw = (const v16h*)(wih + ((size_t)(kt * 16 + nt)) * 512 + lane * 16);
                acc = __builtin_amdgcn_wmma_f32_16x16x32_f16(false, ax[kt], false, *bw,
                                                             (short)0, acc, false, false);
            }
#pragma unroll
            for (int kt = 0; kt < 2; ++kt) {
                const v16h* bw = (const v16h*)(whh + ((size_t)(kt * 16 + nt)) * 512 + lane * 16);
                acc = __builtin_amdgcn_wmma_f32_16x16x32_f16(false, ah[kt], false, *bw,
                                                             (short)0, acc, false, false);
            }
            const int n = nt * 16 + lm;
            const float bs = bias[n];
#pragma unroll
            for (int r = 0; r < 8; ++r)
                gates[r + 8 * hi][n] = acc[r] + bs;
        }
        __syncthreads();
        // elementwise gate update (i,f,g,o order)
        for (int i = lane; i < 16 * OO; i += 32) {
            int r = i >> 6, j = i & 63;
            float ig = 1.0f / (1.0f + __expf(-gates[r][j]));
            float fg = 1.0f / (1.0f + __expf(-gates[r][OO + j]));
            float gg = tanhf(gates[r][2 * OO + j]);
            float og = 1.0f / (1.0f + __expf(-gates[r][3 * OO + j]));
            float c = fg * cbuf[r][j] + ig * gg;
            cbuf[r][j] = c;
            float h = og * tanhf(c);
            hbuf[r][j] = h;
            int nd = base + r;
            if (nd < Bn) {
                float o = relu_out ? (h > 0.0f ? h : 0.0f) : h;
                out[((size_t)nd * TT + t) * OO + j] = o;
            }
        }
        __syncthreads();
    }
}

// ---------------------------------------------------------------------------
extern "C" void kernel_launch(void* const* d_in, const int* in_sizes, int n_in,
                              void* d_out, int out_size, void* d_ws, size_t ws_size,
                              hipStream_t stream) {
    const float* x_node = (const float*)d_in[0];
    const float* x_pod  = (const float*)d_in[1];
    const float* x_svc  = (const float*)d_in[2];
    const int* sc_src = (const int*)d_in[3];  const int* sc_dst = (const int*)d_in[4];
    const int* in_src = (const int*)d_in[5];  const int* in_dst = (const int*)d_in[6];
    const int* ni_src = (const int*)d_in[7];  const int* ni_dst = (const int*)d_in[8];
    const int* ii_src = (const int*)d_in[9];  const int* ii_dst = (const int*)d_in[10];
    const int* si_src = (const int*)d_in[11]; const int* si_dst = (const int*)d_in[12];
    const int* is_src = (const int*)d_in[13]; const int* is_dst = (const int*)d_in[14];
    const float* W_sc = (const float*)d_in[15]; const float* b_sc = (const float*)d_in[16];
    const float* W_in = (const float*)d_in[17]; const float* b_in = (const float*)d_in[18];
    const float* W_ni = (const float*)d_in[19]; const float* b_ni = (const float*)d_in[20];
    const float* W_ii = (const float*)d_in[21]; const float* b_ii = (const float*)d_in[22];
    const float* W_si = (const float*)d_in[23]; const float* b_si = (const float*)d_in[24];
    const float* W_is = (const float*)d_in[25]; const float* b_is = (const float*)d_in[26];
    const float* Wih0 = (const float*)d_in[27]; const float* Whh0 = (const float*)d_in[28];
    const float* bih0 = (const float*)d_in[29]; const float* bhh0 = (const float*)d_in[30];
    const float* Wih1 = (const float*)d_in[31]; const float* Whh1 = (const float*)d_in[32];
    const float* bih1 = (const float*)d_in[33]; const float* bhh1 = (const float*)d_in[34];

    const int E_sc = in_sizes[3], E_in = in_sizes[5], E_ni = in_sizes[7];
    const int E_ii = in_sizes[9], E_si = in_sizes[11], E_is = in_sizes[13];

    size_t cur = 0;
    auto alloc = [&](size_t bytes) -> char* {
        cur = (cur + 255) & ~(size_t)255;
        char* p = (char*)d_ws + cur;
        cur += bytes;
        return p;
    };
    // ---- zeroed region: rsqrt-degree accumulators + scatter targets ----
    float* rvo_sc = (float*)alloc(N_SVC  * 4);
    float* rvo_in = (float*)alloc(N_POD  * 4);
    float* rvo_ni = (float*)alloc(N_NODE * 4);
    float* rvo_ii = (float*)alloc(N_POD  * 4);
    float* rvo_si = (float*)alloc(N_SVC  * 4);
    float* rvo_is = (float*)alloc(N_POD  * 4);
    float* rvi_sc = (float*)alloc(N_SVC  * 4);
    float* rvi_in = (float*)alloc(N_NODE * 4);
    float* rvi_ni = (float*)alloc(N_POD  * 4);
    float* rvi_ii = (float*)alloc(N_POD  * 4);
    float* rvi_si = (float*)alloc(N_POD  * 4);
    float* rvi_is = (float*)alloc(N_SVC  * 4);
    float* m_sc = (float*)alloc((size_t)N_SVC  * 1024 * 4);
    float* m_in = (float*)alloc((size_t)N_NODE * 1024 * 4);
    float* m_ni = (float*)alloc((size_t)N_POD  * 1024 * 4);
    float* m_ii = (float*)alloc((size_t)N_POD  * 1024 * 4);
    float* m_si = (float*)alloc((size_t)N_POD  * 1024 * 4);
    float* m_is = (float*)alloc((size_t)N_SVC  * 1024 * 4);
    size_t zero_bytes = cur;
    // ---- non-zeroed scratch (fully overwritten) ----
    float* h0    = (float*)alloc((size_t)N_TOT * TT * OO * 4);
    float* bias0 = (float*)alloc(4 * OO * 4);
    float* bias1 = (float*)alloc(4 * OO * 4);
    _Float16* pw_sc = (_Float16*)alloc(8192 * 2);
    _Float16* pw_in = (_Float16*)alloc(8192 * 2);
    _Float16* pw_ni = (_Float16*)alloc(8192 * 2);
    _Float16* pw_ii = (_Float16*)alloc(8192 * 2);
    _Float16* pw_si = (_Float16*)alloc(8192 * 2);
    _Float16* pw_is = (_Float16*)alloc(8192 * 2);
    _Float16* pwih0 = (_Float16*)alloc(32768 * 2);
    _Float16* pwhh0 = (_Float16*)alloc(16384 * 2);
    _Float16* pwih1 = (_Float16*)alloc(16384 * 2);
    _Float16* pwhh1 = (_Float16*)alloc(16384 * 2);

    hipMemsetAsync(d_ws, 0, zero_bytes, stream);

    // degrees
    degree_count_k<<<(E_sc + 255) / 256, 256, 0, stream>>>(sc_src, sc_dst, rvo_sc, rvi_sc, E_sc);
    degree_count_k<<<(E_in + 255) / 256, 256, 0, stream>>>(in_src, in_dst, rvo_in, rvi_in, E_in);
    degree_count_k<<<(E_ni + 255) / 256, 256, 0, stream>>>(ni_src, ni_dst, rvo_ni, rvi_ni, E_ni);
    degree_count_k<<<(E_ii + 255) / 256, 256, 0, stream>>>(ii_src, ii_dst, rvo_ii, rvi_ii, E_ii);
    degree_count_k<<<(E_si + 255) / 256, 256, 0, stream>>>(si_src, si_dst, rvo_si, rvi_si, E_si);
    degree_count_k<<<(E_is + 255) / 256, 256, 0, stream>>>(is_src, is_dst, rvo_is, rvi_is, E_is);
    const struct { float* p; int n; } rinvs[12] = {
        {rvo_sc, N_SVC}, {rvo_in, N_POD}, {rvo_ni, N_NODE}, {rvo_ii, N_POD},
        {rvo_si, N_SVC}, {rvo_is, N_POD}, {rvi_sc, N_SVC},  {rvi_in, N_NODE},
        {rvi_ni, N_POD}, {rvi_ii, N_POD}, {rvi_si, N_POD},  {rvi_is, N_SVC}};
    for (int i = 0; i < 12; ++i)
        to_rinv_k<<<(rinvs[i].n + 255) / 256, 256, 0, stream>>>(rinvs[i].p, rinvs[i].n);

    // pack weights to f16 fragment layout
    pack_b16_k<<<(8192 + 255) / 256, 256, 0, stream>>>(W_sc, pw_sc, FF, HH, 0);
    pack_b16_k<<<(8192 + 255) / 256, 256, 0, stream>>>(W_in, pw_in, FF, HH, 0);
    pack_b16_k<<<(8192 + 255) / 256, 256, 0, stream>>>(W_ni, pw_ni, FF, HH, 0);
    pack_b16_k<<<(8192 + 255) / 256, 256, 0, stream>>>(W_ii, pw_ii, FF, HH, 0);
    pack_b16_k<<<(8192 + 255) / 256, 256, 0, stream>>>(W_si, pw_si, FF, HH, 0);
    pack_b16_k<<<(8192 + 255) / 256, 256, 0, stream>>>(W_is, pw_is, FF, HH, 0);
    pack_b16_k<<<(32768 + 255) / 256, 256, 0, stream>>>(Wih0, pwih0, HH, 4 * OO, 1);
    pack_b16_k<<<(16384 + 255) / 256, 256, 0, stream>>>(Whh0, pwhh0, OO, 4 * OO, 1);
    pack_b16_k<<<(16384 + 255) / 256, 256, 0, stream>>>(Wih1, pwih1, OO, 4 * OO, 1);
    pack_b16_k<<<(16384 + 255) / 256, 256, 0, stream>>>(Whh1, pwhh1, OO, 4 * OO, 1);
    add_bias_k<<<1, 256, 0, stream>>>(bih0, bhh0, bias0, 4 * OO);
    add_bias_k<<<1, 256, 0, stream>>>(bih1, bhh1, bias1, 4 * OO);

    // edge scatter (src-scaled), native f32 atomics
    scatter_add_k<<<E_sc, 256, 0, stream>>>(x_svc,  sc_src, sc_dst, rvo_sc, m_sc);
    scatter_add_k<<<E_in, 256, 0, stream>>>(x_pod,  in_src, in_dst, rvo_in, m_in);
    scatter_add_k<<<E_ni, 256, 0, stream>>>(x_node, ni_src, ni_dst, rvo_ni, m_ni);
    scatter_add_k<<<E_ii, 256, 0, stream>>>(x_pod,  ii_src, ii_dst, rvo_ii, m_ii);
    scatter_add_k<<<E_si, 256, 0, stream>>>(x_svc,  si_src, si_dst, rvo_si, m_si);
    scatter_add_k<<<E_is, 256, 0, stream>>>(x_pod,  is_src, is_dst, rvo_is, m_is);

    // feat = relu(concat(node, pod, svc)) -> d_out[0 : 22500*16*128]
    float* out = (float*)d_out;
    feat_combine_k<<<N_NODE * TT / 16, 32, 0, stream>>>(
        m_in, rvi_in, pw_in, b_in,
        nullptr, nullptr, nullptr, nullptr,
        nullptr, nullptr, nullptr, nullptr,
        1, 1.0f, out);
    feat_combine_k<<<N_POD * TT / 16, 32, 0, stream>>>(
        m_ni, rvi_ni, pw_ni, b_ni,
        m_ii, rvi_ii, pw_ii, b_ii,
        m_si, rvi_si, pw_si, b_si,
        3, 1.0f / 3.0f, out + (size_t)N_NODE * TT * HH);
    feat_combine_k<<<N_SVC * TT / 16, 32, 0, stream>>>(
        m_sc, rvi_sc, pw_sc, b_sc,
        m_is, rvi_is, pw_is, b_is,
        nullptr, nullptr, nullptr, nullptr,
        2, 0.5f, out + (size_t)(N_NODE + N_POD) * TT * HH);

    // 2-layer LSTM; final relu -> d_out[22500*16*128 : ]
    const int nblk = (N_TOT + 15) / 16;
    lstm_layer_k<HH><<<nblk, 32, 0, stream>>>(out, pwih0, pwhh0, bias0, h0, N_TOT, 0);
    lstm_layer_k<OO><<<nblk, 32, 0, stream>>>(h0, pwih1, pwhh1, bias1,
                                              out + (size_t)N_TOT * TT * HH, N_TOT, 1);
}
